// SkimRNN_20547123544846
// MI455X (gfx1250) — compile-verified, matrix operand order checked by reading, not run on hardware
//
#include <hip/hip_runtime.h>
#include <stdint.h>

#define B_   128
#define S_   100
#define D_   512
#define NCD  16     // codes per token
#define GD   2048   // 4*D
#define SM   64
#define GS   256    // 4*SM
#define EPSF 1e-20f

typedef __attribute__((ext_vector_type(16))) __bf16       bf16x16;
typedef __attribute__((ext_vector_type(8)))  float        f32x8;
typedef __attribute__((ext_vector_type(4)))  unsigned int u32x4;
typedef __attribute__((ext_vector_type(8)))  int          i32x8;
typedef __attribute__((ext_vector_type(4)))  int          i32x4;

__device__ __forceinline__ f32x8 wmma_bf16(bf16x16 a, bf16x16 b, f32x8 c) {
  // D = A*B + C, fp32 accumulate  ->  v_wmma_f32_16x16x32_bf16
  return __builtin_amdgcn_wmma_f32_16x16x32_bf16(false, a, false, b, (short)0, c, false, false);
}

__device__ __forceinline__ float sig_(float x) { return 1.0f / (1.0f + __expf(-x)); }

// Load a 16x32 bf16 operand fragment (A or B^T) : lane L -> row r0+(L&15), k chunk by (L>>4)
__device__ __forceinline__ bf16x16 ldfrag(const __bf16* base, int stride, int r0, int k0, int L) {
  return *(const bf16x16*)(base + (size_t)(r0 + (L & 15)) * stride + k0 + ((L >> 4) << 4));
}

// ---- Tensor Data Mover: 2D tile [rows=128, cols=64 bf16] of a [128 x 512] bf16 tensor -> LDS
__device__ __forceinline__ void tdm_load_h(const __bf16* gsrc, unsigned lds_off) {
  unsigned long long ga = (unsigned long long)(uintptr_t)gsrc;
  u32x4 g0;
  g0[0] = 1u;                                             // count = 1, user mode
  g0[1] = lds_off;                                        // lds_addr (bytes)
  g0[2] = (unsigned)(ga & 0xFFFFFFFFull);                 // global_addr[31:0]
  g0[3] = (unsigned)((ga >> 32) & 0x01FFFFFFull) | (2u << 30); // global_addr[56:32] | type=2
  i32x8 g1;
  g1[0] = (int)(1u << 16);        // workgroup_mask=0, data_size=1 (2 bytes)
  g1[1] = (int)(512u << 16);      // tensor_dim0 = 512 (low 16 at [31:16])
  g1[2] = (int)(128u << 16);      // tensor_dim1 = 128 (low 16 at [31:16])
  g1[3] = (int)(64u << 16);       // tile_dim0 = 64
  g1[4] = 128;                    // tile_dim1 = 128
  g1[5] = 512;                    // tensor_dim0_stride = 512 elements
  g1[6] = 0;
  g1[7] = 0;
  i32x4 gz4; gz4[0] = 0; gz4[1] = 0; gz4[2] = 0; gz4[3] = 0;
  i32x8 gz8; gz8[0] = 0; gz8[1] = 0; gz8[2] = 0; gz8[3] = 0;
  gz8[4] = 0; gz8[5] = 0; gz8[6] = 0; gz8[7] = 0;
  __builtin_amdgcn_tensor_load_to_lds(g0, g1, gz4, gz4, gz8, 0);
}

// ---- global software barrier (monotonic counter, all 17 blocks resident) ----
__device__ __forceinline__ void gbarrier(unsigned* cnt, unsigned target) {
  __syncthreads();
  if (threadIdx.x == 0) {
    __hip_atomic_fetch_add(cnt, 1u, __ATOMIC_ACQ_REL, __HIP_MEMORY_SCOPE_AGENT);
    while (__hip_atomic_load(cnt, __ATOMIC_ACQUIRE, __HIP_MEMORY_SCOPE_AGENT) < target)
      __builtin_amdgcn_s_sleep(2);
  }
  __syncthreads();
}

// ================= init: zero state + barrier counter =================
__global__ void k_init(float* hl0, float* cl0, __bf16* hlbf0,
                       float* hsf0, __bf16* hsbf0, float* cs0, unsigned* cnt) {
  size_t i = (size_t)blockIdx.x * blockDim.x + threadIdx.x;
  if (i == 0) *cnt = 0u;
  if (i < (size_t)B_ * D_) { hl0[i] = 0.f; cl0[i] = 0.f; hlbf0[i] = (__bf16)0.0f; }
  if (i < (size_t)B_ * SM) { hsf0[i] = 0.f; hsbf0[i] = (__bf16)0.0f; cs0[i] = 0.f; }
}

// ================= convert weights to bf16, fold biases =================
__global__ void k_cvt(const float* wihl, const float* whhl, const float* wihs, const float* whhs,
                      const float* bihl, const float* bhhl, const float* bihs, const float* bhhs,
                      __bf16* wihl_b, __bf16* whhl_b, __bf16* wihs_b, __bf16* whhs_b,
                      float* bsuml, float* bsums) {
  for (size_t i = (size_t)blockIdx.x * blockDim.x + threadIdx.x;
       i < (size_t)GD * D_; i += (size_t)gridDim.x * blockDim.x) {
    wihl_b[i] = (__bf16)wihl[i];
    whhl_b[i] = (__bf16)whhl[i];
    if (i < (size_t)GS * D_) wihs_b[i] = (__bf16)wihs[i];
    if (i < (size_t)GS * SM) whhs_b[i] = (__bf16)whhs[i];
    if (i < GD)  bsuml[i] = bihl[i] + bhhl[i];
    if (i < GS)  bsums[i] = bihs[i] + bhhs[i];
  }
}

// ========== embedding sum -> e_bf16 [S*B, 512]; fold e-part of gate into ybuf ==========
__global__ void k_embed(const int* __restrict__ x, const float* __restrict__ emb,
                        const float* __restrict__ wgate, const float* __restrict__ bgate,
                        __bf16* __restrict__ ebf, float* __restrict__ ybuf) {
  const int m = blockIdx.x;          // m = t*B + b
  const int t = m >> 7, b = m & 127;
  __shared__ int   sidx[NCD];
  __shared__ float red[2][128];
  if (threadIdx.x < NCD) sidx[threadIdx.x] = x[(size_t)b * S_ * NCD + (size_t)t * NCD + threadIdx.x];
  __syncthreads();
  const int d0 = threadIdx.x * 4;
  float v0 = 0.f, v1 = 0.f, v2 = 0.f, v3 = 0.f;
  for (int n = 0; n < NCD; ++n) {
    const float* e = emb + (size_t)sidx[n] * D_ + d0;   // padding row (VOCAB) is zero in emb
    v0 += e[0]; v1 += e[1]; v2 += e[2]; v3 += e[3];
  }
  __bf16* eo = ebf + (size_t)m * D_ + d0;
  eo[0] = (__bf16)v0; eo[1] = (__bf16)v1; eo[2] = (__bf16)v2; eo[3] = (__bf16)v3;
  float p0 = v0 * wgate[d0] + v1 * wgate[d0 + 1] + v2 * wgate[d0 + 2] + v3 * wgate[d0 + 3];
  float p1 = v0 * wgate[1536 + d0] + v1 * wgate[1536 + d0 + 1] +
             v2 * wgate[1536 + d0 + 2] + v3 * wgate[1536 + d0 + 3];
  red[0][threadIdx.x] = p0; red[1][threadIdx.x] = p1;
  __syncthreads();
  for (int s = 64; s > 0; s >>= 1) {
    if (threadIdx.x < (unsigned)s) {
      red[0][threadIdx.x] += red[0][threadIdx.x + s];
      red[1][threadIdx.x] += red[1][threadIdx.x + s];
    }
    __syncthreads();
  }
  if (threadIdx.x == 0) {
    ybuf[(size_t)m * 2 + 0] = red[0][0] + bgate[0];
    ybuf[(size_t)m * 2 + 1] = red[1][0] + bgate[1];
  }
}

// ========== batched input GEMM: out[M,N] = A[M,512] * W[N,512]^T + bias[N] ==========
__global__ void __launch_bounds__(256) k_xgemm(const __bf16* __restrict__ A,
                                               const __bf16* __restrict__ W,
                                               const float* __restrict__ bias,
                                               float* __restrict__ out, int N) {
  const int wid = threadIdx.x >> 5, L = threadIdx.x & 31;
  const int m0 = blockIdx.x * 128 + (wid >> 1) * 32;
  const int n0 = blockIdx.y * 128 + (wid & 1) * 64;
  f32x8 acc[2][4] = {};
  for (int k0 = 0; k0 < D_; k0 += 32) {
    bf16x16 a0 = ldfrag(A, D_, m0,      k0, L);
    bf16x16 a1 = ldfrag(A, D_, m0 + 16, k0, L);
#pragma unroll
    for (int nt = 0; nt < 4; ++nt) {
      bf16x16 bfr = ldfrag(W, D_, n0 + nt * 16, k0, L);
      acc[0][nt] = wmma_bf16(a0, bfr, acc[0][nt]);
      acc[1][nt] = wmma_bf16(a1, bfr, acc[1][nt]);
    }
  }
#pragma unroll
  for (int nt = 0; nt < 4; ++nt) {
    const int col = n0 + nt * 16 + (L & 15);
    const float bs = bias[col];
#pragma unroll
    for (int mt = 0; mt < 2; ++mt) {
#pragma unroll
      for (int r = 0; r < 8; ++r) {
        const int row = m0 + mt * 16 + ((L >> 4) << 3) + r;
        out[(size_t)row * N + col] = acc[mt][nt][r] + bs;
      }
    }
  }
}

// ========== persistent scan: 16 WGs = large cell (j-partitioned), WG 16 = small cell ==========
__global__ void __launch_bounds__(256) k_scan(
    const float* __restrict__ Gxl, const float* __restrict__ Gxs,
    const __bf16* __restrict__ whhl_b, const __bf16* __restrict__ whhs_b,
    const float* __restrict__ wgate, const float* __restrict__ gum,
    float* __restrict__ ybuf,
    float* hl0, float* hl1, float* cl0, float* cl1,
    __bf16* hlbf0, __bf16* hlbf1,
    float* hl_pre, float* cl_pre,
    float* hsf0, float* hsf1, __bf16* hsbf0, __bf16* hsbf1,
    float* cs0, float* cs1, unsigned* cnt) {
  __shared__ __bf16 smA[2][B_ * 64];  // 2 x 16KB: double-buffered K-chunks of h (TDM DMA)
  const int wid = threadIdx.x >> 5, L = threadIdx.x & 31;
  const int wg = blockIdx.x;
  const int j0 = wg * 32;
  const int m0 = (wid >> 1) * 32;       // large cell: 4 M groups x 2 N groups
  const int jb = j0 + (wid & 1) * 16;
  const int m0s = wid * 16;             // small cell: 8 waves x 16 rows
  unsigned bar = 0;

  for (int t = 0; t < S_; ++t) {
    const int cur = t & 1;
    const float*  hl_cur   = cur ? hl1 : hl0;
    float*        hl_nxt   = cur ? hl0 : hl1;
    const float*  cl_cur   = cur ? cl1 : cl0;
    float*        cl_nxt   = cur ? cl0 : cl1;
    const __bf16* hlbf_cur = cur ? hlbf1 : hlbf0;
    __bf16*       hlbf_nxt = cur ? hlbf0 : hlbf1;
    const __bf16* hsbf_cur = cur ? hsbf1 : hsbf0;
    __bf16*       hsbf_nxt = cur ? hsbf0 : hsbf1;
    float*        hsf_nxt  = cur ? hsf0 : hsf1;
    const float*  cs_cur   = cur ? cs1 : cs0;
    float*        cs_nxt   = cur ? cs0 : cs1;

    // ---------------- phase 1 ----------------
    if (wg < 16) {
      // Prefetch this step's Gx gate slab (128 rows x 4 gate segments of 32 floats)
      // while the WMMA pipeline below runs.  -> global_prefetch_b8
      {
        const float* gxl0 = Gxl + (size_t)t * B_ * GD;
        int seg0 = threadIdx.x;             // 512 segments of 128B
        __builtin_prefetch(gxl0 + (size_t)(seg0 & 127) * GD + (seg0 >> 7) * 512 + j0, 0, 1);
        int seg1 = threadIdx.x + 256;
        __builtin_prefetch(gxl0 + (size_t)(seg1 & 127) * GD + (seg1 >> 7) * 512 + j0, 0, 1);
      }
      f32x8 acc[2][4] = {};
      // prologue: DMA chunk 0 into buffer 0
      if (threadIdx.x == 0) tdm_load_h(hlbf_cur, (unsigned)(uintptr_t)&smA[0][0]);
      for (int kc = 0; kc < 8; ++kc) {             // K = 8 chunks of 64
        __syncthreads();   // all waves done with buffer (kc+1)&1 from chunk kc-1
        if (threadIdx.x == 0) {
          if (kc < 7) {
            // prefetch next chunk into the other buffer, then wait only for chunk kc
            tdm_load_h(hlbf_cur + (kc + 1) * 64, (unsigned)(uintptr_t)&smA[(kc + 1) & 1][0]);
            __builtin_amdgcn_s_wait_tensorcnt(1);
          } else {
            __builtin_amdgcn_s_wait_tensorcnt(0);
          }
        }
        __syncthreads();
        const __bf16* smc = &smA[kc & 1][0];
#pragma unroll
        for (int ki = 0; ki < 2; ++ki) {
          const int lk0 = ki * 32, kg = kc * 64 + lk0;
          bf16x16 a0 = ldfrag(smc, 64, m0,      lk0, L);
          bf16x16 a1 = ldfrag(smc, 64, m0 + 16, lk0, L);
#pragma unroll
          for (int g = 0; g < 4; ++g) {
            bf16x16 bfr = ldfrag(whhl_b, D_, g * 512 + jb, kg, L);
            acc[0][g] = wmma_bf16(a0, bfr, acc[0][g]);
            acc[1][g] = wmma_bf16(a1, bfr, acc[1][g]);
          }
        }
      }
      const float* gxl = Gxl + (size_t)t * B_ * GD;
#pragma unroll
      for (int mt = 0; mt < 2; ++mt) {
#pragma unroll
        for (int r = 0; r < 8; ++r) {
          const int bb = m0 + mt * 16 + ((L >> 4) << 3) + r;
          const int jj = jb + (L & 15);
          const size_t gb = (size_t)bb * GD;
          float iv = acc[mt][0][r] + gxl[gb + jj];
          float fv = acc[mt][1][r] + gxl[gb + 512 + jj];
          float gv = acc[mt][2][r] + gxl[gb + 1024 + jj];
          float ov = acc[mt][3][r] + gxl[gb + 1536 + jj];
          const size_t idx = (size_t)bb * D_ + jj;
          float cn = sig_(fv) * cl_cur[idx] + sig_(iv) * tanhf(gv);
          float hn = sig_(ov) * tanhf(cn);
          hl_pre[idx] = hn; cl_pre[idx] = cn;
          float p0 = wgate[512 + jj] * hn + wgate[1024 + jj] * cn;
          float p1 = wgate[1536 + 512 + jj] * hn + wgate[1536 + 1024 + jj] * cn;
#pragma unroll
          for (int o = 1; o < 16; o <<= 1) {
            p0 += __shfl_xor(p0, o, 16);
            p1 += __shfl_xor(p1, o, 16);
          }
          if ((L & 15) == 0) {
            atomicAdd(&ybuf[((size_t)t * B_ + bb) * 2 + 0], p0);
            atomicAdd(&ybuf[((size_t)t * B_ + bb) * 2 + 1], p1);
          }
        }
      }
    } else {
      // small LSTM: [128,256] = h_s[128,64] @ w_hh_s[256,64]^T  (+ Gxs)
      f32x8 acc[16] = {};
#pragma unroll
      for (int ki = 0; ki < 2; ++ki) {
        const int k0 = ki * 32;
        bf16x16 a = ldfrag(hsbf_cur, SM, m0s, k0, L);
#pragma unroll
        for (int nt = 0; nt < 16; ++nt) {
          bf16x16 bfr = ldfrag(whhs_b, SM, nt * 16, k0, L);
          acc[nt] = wmma_bf16(a, bfr, acc[nt]);
        }
      }
      const float* gxs = Gxs + (size_t)t * B_ * GS;
#pragma unroll
      for (int jt = 0; jt < 4; ++jt) {
#pragma unroll
        for (int r = 0; r < 8; ++r) {
          const int bb = m0s + ((L >> 4) << 3) + r;
          const int jj = jt * 16 + (L & 15);
          const size_t gb = (size_t)bb * GS;
          float iv = acc[jt][r]      + gxs[gb + jj];
          float fv = acc[4 + jt][r]  + gxs[gb + 64 + jj];
          float gv = acc[8 + jt][r]  + gxs[gb + 128 + jj];
          float ov = acc[12 + jt][r] + gxs[gb + 192 + jj];
          float cn = sig_(fv) * cs_cur[bb * SM + jj] + sig_(iv) * tanhf(gv);
          float hn = sig_(ov) * tanhf(cn);
          cs_nxt[bb * SM + jj] = cn;
          hsf_nxt[bb * SM + jj] = hn;
          hsbf_nxt[bb * SM + jj] = (__bf16)hn;
        }
      }
    }
    bar += 17; gbarrier(cnt, bar);

    // ---------------- phase 2: gumbel-softmax mix ----------------
    if (wg < 16) {
#pragma unroll
      for (int mt = 0; mt < 2; ++mt) {
#pragma unroll
        for (int r = 0; r < 8; ++r) {
          const int bb = m0 + mt * 16 + ((L >> 4) << 3) + r;
          const int jj = jb + (L & 15);
          const size_t yb = ((size_t)t * B_ + bb) * 2;
          float gy0 = ybuf[yb]     - logf(-logf(gum[yb]     + EPSF) + EPSF);
          float gy1 = ybuf[yb + 1] - logf(-logf(gum[yb + 1] + EPSF) + EPSF);
          float r1v = 1.0f / (1.0f + __expf(gy0 - gy1));
          float r0v = 1.0f - r1v;
          const size_t idx = (size_t)bb * D_ + jj;
          float hp = hl_pre[idx], cp = cl_pre[idx];
          float hmx = (jj < SM) ? hsf_nxt[bb * SM + jj] : hl_cur[idx];
          float cmx = (jj < SM) ? cs_nxt[bb * SM + jj]  : cp;
          float hn = r0v * hp + r1v * hmx;
          float cn = r0v * cp + r1v * cmx;
          hl_nxt[idx] = hn; cl_nxt[idx] = cn; hlbf_nxt[idx] = (__bf16)hn;
        }
      }
    }
    bar += 17; gbarrier(cnt, bar);
  }
}

// ========== classifier: out = relu(h) @ w_cls^T + b_cls ==========
__global__ void k_cls(const float* __restrict__ hl, const float* __restrict__ wcls,
                      const float* __restrict__ bcls, float* __restrict__ out) {
  __shared__ float red[2][128];
  const int b = blockIdx.x, tid = threadIdx.x;
  float p0 = 0.f, p1 = 0.f;
  for (int d = tid; d < D_; d += 128) {
    float v = hl[(size_t)b * D_ + d];
    v = v > 0.f ? v : 0.f;
    p0 += v * wcls[d]; p1 += v * wcls[D_ + d];
  }
  red[0][tid] = p0; red[1][tid] = p1;
  __syncthreads();
  for (int s = 64; s > 0; s >>= 1) {
    if (tid < s) { red[0][tid] += red[0][tid + s]; red[1][tid] += red[1][tid + s]; }
    __syncthreads();
  }
  if (tid == 0) {
    out[b * 2 + 0] = red[0][0] + bcls[0];
    out[b * 2 + 1] = red[1][0] + bcls[1];
  }
}

extern "C" void kernel_launch(void* const* d_in, const int* in_sizes, int n_in,
                              void* d_out, int out_size, void* d_ws, size_t ws_size,
                              hipStream_t stream) {
  (void)in_sizes; (void)n_in; (void)out_size; (void)ws_size;
  const int*   x     = (const int*)d_in[0];
  const float* gum   = (const float*)d_in[5];
  const float* emb   = (const float*)d_in[6];
  const float* wihl  = (const float*)d_in[7];
  const float* whhl  = (const float*)d_in[8];
  const float* bihl  = (const float*)d_in[9];
  const float* bhhl  = (const float*)d_in[10];
  const float* wihs  = (const float*)d_in[11];
  const float* whhs  = (const float*)d_in[12];
  const float* bihs  = (const float*)d_in[13];
  const float* bhhs  = (const float*)d_in[14];
  const float* wgate = (const float*)d_in[15];
  const float* bgate = (const float*)d_in[16];
  const float* wcls  = (const float*)d_in[17];
  const float* bcls  = (const float*)d_in[18];
  float* out = (float*)d_out;

  char* base = (char*)d_ws;
  size_t off = 0;
  auto alloc = [&](size_t bytes) -> char* {
    char* r = base + off;
    off = (off + bytes + 255) & ~(size_t)255;
    return r;
  };
  __bf16* ebf    = (__bf16*)alloc((size_t)S_ * B_ * D_ * 2);
  float*  Gxl    = (float*)alloc((size_t)S_ * B_ * GD * 4);
  float*  Gxs    = (float*)alloc((size_t)S_ * B_ * GS * 4);
  __bf16* wihl_b = (__bf16*)alloc((size_t)GD * D_ * 2);
  __bf16* whhl_b = (__bf16*)alloc((size_t)GD * D_ * 2);
  __bf16* wihs_b = (__bf16*)alloc((size_t)GS * D_ * 2);
  __bf16* whhs_b = (__bf16*)alloc((size_t)GS * SM * 2);
  float*  bsuml  = (float*)alloc(GD * 4);
  float*  bsums  = (float*)alloc(GS * 4);
  float*  ybuf   = (float*)alloc((size_t)S_ * B_ * 2 * 4);
  float*  hl0    = (float*)alloc((size_t)B_ * D_ * 4);
  float*  hl1    = (float*)alloc((size_t)B_ * D_ * 4);
  float*  cl0    = (float*)alloc((size_t)B_ * D_ * 4);
  float*  cl1    = (float*)alloc((size_t)B_ * D_ * 4);
  __bf16* hlbf0  = (__bf16*)alloc((size_t)B_ * D_ * 2);
  __bf16* hlbf1  = (__bf16*)alloc((size_t)B_ * D_ * 2);
  float*  hl_pre = (float*)alloc((size_t)B_ * D_ * 4);
  float*  cl_pre = (float*)alloc((size_t)B_ * D_ * 4);
  float*  hsf0   = (float*)alloc((size_t)B_ * SM * 4);
  float*  hsf1   = (float*)alloc((size_t)B_ * SM * 4);
  __bf16* hsbf0  = (__bf16*)alloc((size_t)B_ * SM * 2);
  __bf16* hsbf1  = (__bf16*)alloc((size_t)B_ * SM * 2);
  float*  cs0    = (float*)alloc((size_t)B_ * SM * 4);
  float*  cs1    = (float*)alloc((size_t)B_ * SM * 4);
  unsigned* cnt  = (unsigned*)alloc(256);

  k_init<<<256, 256, 0, stream>>>(hl0, cl0, hlbf0, hsf0, hsbf0, cs0, cnt);
  k_cvt<<<512, 256, 0, stream>>>(wihl, whhl, wihs, whhs, bihl, bhhl, bihs, bhhs,
                                 wihl_b, whhl_b, wihs_b, whhs_b, bsuml, bsums);
  k_embed<<<S_ * B_, 128, 0, stream>>>(x, emb, wgate, bgate, ebf, ybuf);
  k_xgemm<<<dim3(S_, 16), 256, 0, stream>>>(ebf, wihl_b, bsuml, Gxl, GD);
  k_xgemm<<<dim3(S_, 2), 256, 0, stream>>>(ebf, wihs_b, bsums, Gxs, GS);
  k_scan<<<17, 256, 0, stream>>>(Gxl, Gxs, whhl_b, whhs_b, wgate, gum, ybuf,
                                 hl0, hl1, cl0, cl1, hlbf0, hlbf1, hl_pre, cl_pre,
                                 hsf0, hsf1, hsbf0, hsbf1, cs0, cs1, cnt);
  // 100 steps: reads buf[t&1], writes buf[(t+1)&1] -> final state in buffer 0
  k_cls<<<B_, 128, 0, stream>>>(hl0, wcls, bcls, out);
}
